// DirectionClassifier_34471407517778
// MI455X (gfx1250) — matrix operand order repaired and checked by reading, not compiled, not run on hardware
//
#include <hip/hip_runtime.h>
#include <hip/hip_bf16.h>

typedef __attribute__((ext_vector_type(16))) _Float16 v16h;
typedef __attribute__((ext_vector_type(8)))  float    v8f;
typedef __attribute__((ext_vector_type(4)))  uint32_t u32x4;
typedef __attribute__((ext_vector_type(8)))  uint32_t u32x8;

// Problem constants
#define BB   32
#define LL   256
#define FF   20
#define HH   256
#define DI   512
#define NN   64
#define KK   4
#define RR   16
#define RC   144          // R + 2N
#define BL   (BB*LL)      // 8192

// LDS byte offset of a shared-memory object (generic -> addrspace(3) -> int)
__device__ __forceinline__ uint32_t ldsoff(const void* p) {
  return (uint32_t)(uintptr_t)(__attribute__((address_space(3))) const void*)p;
}

// Per-lane async DMA: global (saddr base + 32-bit voffset) -> LDS byte offset.
__device__ __forceinline__ void async_g2lds_b32(uint32_t lds_byte,
                                                uint32_t goff_byte,
                                                const void* sbase) {
  asm volatile("global_load_async_to_lds_b32 %0, %1, %2"
               :: "v"(lds_byte), "v"(goff_byte), "s"(sbase)
               : "memory");
}

__device__ __forceinline__ void wait_asynccnt0() {
#if __has_builtin(__builtin_amdgcn_s_wait_asynccnt)
  __builtin_amdgcn_s_wait_asynccnt(0);
#else
  asm volatile("s_wait_asynccnt 0x0" ::: "memory");
#endif
}

// ---------------------------------------------------------------------------
// Fold emb (H,F) + in_proj u-half (DI,H) into Wc (F,DI) and bias (DI).
// ---------------------------------------------------------------------------
__global__ __launch_bounds__(256)
void prep_kernel(const float* __restrict__ ew, const float* __restrict__ eb,
                 const float* __restrict__ ipw, const float* __restrict__ ipb,
                 float* __restrict__ Wc, float* __restrict__ Bc) {
  int idx = blockIdx.x * 256 + threadIdx.x;
  if (idx < FF * DI) {
    int f = idx / DI, d = idx % DI;
    const float* wrow = ipw + d * HH;           // rows 0..DI-1 = u half
    float acc = 0.f;
    for (int h = 0; h < HH; ++h) acc += ew[h * FF + f] * wrow[h];
    Wc[f * DI + d] = acc;
  } else if (idx < FF * DI + DI) {
    int d = idx - FF * DI;
    const float* wrow = ipw + d * HH;
    float acc = ipb[d];
    for (int h = 0; h < HH; ++h) acc += eb[h] * wrow[h];
    Bc[d] = acc;
  }
}

// h at last timestep only: (B,H)
__global__ __launch_bounds__(256)
void hlast_kernel(const float* __restrict__ x, const float* __restrict__ ew,
                  const float* __restrict__ eb, float* __restrict__ Hl) {
  int idx = blockIdx.x * 256 + threadIdx.x;     // B*H
  int b = idx >> 8, hh = idx & 255;
  const float* xr = x + (b * LL + (LL - 1)) * FF;
  float acc = eb[hh];
  for (int f = 0; f < FF; ++f) acc += xr[f] * ew[hh * FF + f];
  Hl[idx] = acc;
}

// z at last timestep only: (B,DI), uses in_proj rows DI..2DI-1
__global__ __launch_bounds__(256)
void zlast_kernel(const float* __restrict__ Hl, const float* __restrict__ ipw,
                  const float* __restrict__ ipb, float* __restrict__ Zl) {
  int idx = blockIdx.x * 256 + threadIdx.x;     // B*DI
  int b = idx >> 9, d = idx & 511;
  const float* hr = Hl + b * HH;
  const float* wrow = ipw + (DI + d) * HH;
  float acc = ipb[DI + d];
  for (int h = 0; h < HH; ++h) acc += hr[h] * wrow[h];
  Zl[idx] = acc;
}

// ---------------------------------------------------------------------------
// WMMA GEMM 1: Upre(BL,DI) = X(BL,F) @ Wc(F,DI) + bias   (K=20 padded to 32)
// ---------------------------------------------------------------------------
__global__ __launch_bounds__(256)
void gemm_upre_wmma(const float* __restrict__ X, const float* __restrict__ Wc,
                    const float* __restrict__ Bc, float* __restrict__ Upre) {
  const int lane = threadIdx.x & 31;
  const int wave = threadIdx.x >> 5;
  const int tile = blockIdx.x * 8 + wave;       // (BL/16)*(DI/16) = 16384
  const int mT = tile >> 5;
  const int nT = tile & 31;
  const int half = lane >> 4;
  const int l16  = lane & 15;

  const int m = mT * 16 + l16;
  const float* xr = X + m * FF;
  const int kbA = half * 16;
  v16h a;
#pragma unroll
  for (int j = 0; j < 4; ++j) {
    int k0 = kbA + 2 * j;
    a[2 * j]     = (k0     < FF) ? (_Float16)xr[k0]     : (_Float16)0.f;
    a[2 * j + 1] = (k0 + 1 < FF) ? (_Float16)xr[k0 + 1] : (_Float16)0.f;
    int k1 = 8 + kbA + 2 * j;
    a[2 * j + 8] = (k1     < FF) ? (_Float16)xr[k1]     : (_Float16)0.f;
    a[2 * j + 9] = (k1 + 1 < FF) ? (_Float16)xr[k1 + 1] : (_Float16)0.f;
  }

  const int n = nT * 16 + l16;
  const int kbB = half * 16;
  v16h bm;
#pragma unroll
  for (int j = 0; j < 8; ++j) {
    int k = kbB + 2 * j;
    bm[2 * j]     = (k     < FF) ? (_Float16)Wc[k * DI + n]       : (_Float16)0.f;
    bm[2 * j + 1] = (k + 1 < FF) ? (_Float16)Wc[(k + 1) * DI + n] : (_Float16)0.f;
  }

  v8f c = {};
  c = __builtin_amdgcn_wmma_f32_16x16x32_f16(false, a, false, bm, (short)0, c,
                                             false, false);
  const float bias = Bc[n];
#pragma unroll
  for (int r = 0; r < 8; ++r) {
    int row = mT * 16 + half * 8 + r;
    Upre[row * DI + n] = c[r] + bias;
  }
}

// ---------------------------------------------------------------------------
// Causal depthwise conv (K=4) + SiLU, elementwise over (BL,DI)
// ---------------------------------------------------------------------------
__global__ __launch_bounds__(256)
void conv_silu_kernel(const float* __restrict__ Up, const float* __restrict__ cw,
                      const float* __restrict__ cb, float* __restrict__ Uo) {
  int idx = blockIdx.x * 256 + threadIdx.x;     // BL*DI
  int d  = idx & 511;
  int bt = idx >> 9;
  int t  = bt & 255;
  float acc = cb[d];
#pragma unroll
  for (int k = 0; k < KK; ++k) {
    int tt = t - (KK - 1) + k;
    float v = (tt >= 0) ? Up[(bt - (KK - 1) + k) * DI + d] : 0.f;
    acc += v * cw[d * KK + k];
  }
  Uo[idx] = acc / (1.f + __expf(-acc));         // SiLU
}

// ---------------------------------------------------------------------------
// WMMA GEMM 2: DBC(BL,144) = U(BL,DI) @ x_proj_w^T   (K=512, unroll 4)
// ---------------------------------------------------------------------------
__global__ __launch_bounds__(256)
void gemm_dbc_wmma(const float* __restrict__ U, const float* __restrict__ XPW,
                   float* __restrict__ DBC) {
  const int lane = threadIdx.x & 31;
  const int wave = threadIdx.x >> 5;
  const int tile = blockIdx.x * 8 + wave;       // (BL/16)*(144/16) = 4608
  const int mT = tile / 9;
  const int nT = tile % 9;
  const int half = lane >> 4;
  const int l16  = lane & 15;

  const int m = mT * 16 + l16;
  const int n = nT * 16 + l16;
  const float* ur = U + m * DI;
  const float* wr = XPW + n * DI;               // row n of x_proj_w, contiguous

  v8f c = {};
#pragma unroll 4
  for (int kk = 0; kk < DI; kk += 32) {
    v16h a, bm;
    const int kbA = kk + half * 16;
#pragma unroll
    for (int j = 0; j < 4; ++j) {
      a[2 * j]     = (_Float16)ur[kbA + 2 * j];
      a[2 * j + 1] = (_Float16)ur[kbA + 2 * j + 1];
      a[2 * j + 8] = (_Float16)ur[8 + kbA + 2 * j];
      a[2 * j + 9] = (_Float16)ur[8 + kbA + 2 * j + 1];
    }
    const int kbB = kk + half * 16;
#pragma unroll
    for (int j = 0; j < 8; ++j) {
      bm[2 * j]     = (_Float16)wr[kbB + 2 * j];
      bm[2 * j + 1] = (_Float16)wr[kbB + 2 * j + 1];
    }
    c = __builtin_amdgcn_wmma_f32_16x16x32_f16(false, a, false, bm, (short)0, c,
                                               false, false);
  }
#pragma unroll
  for (int r = 0; r < 8; ++r) {
    int row = mT * 16 + half * 8 + r;
    DBC[row * RC + n] = c[r];
  }
}

// ---------------------------------------------------------------------------
// Selective scan. One thread per (b,d). Per 16-timestep chunk:
//  - TDM tensor_load_to_lds stages the (16 x 80) dt/Bc tile (row stride 144)
//  - global_load_async_to_lds_b32 stages the (16 x 256) U tile
// Waits: s_wait_tensorcnt / s_wait_asynccnt + workgroup barrier.
// ---------------------------------------------------------------------------
__global__ __launch_bounds__(256)
void scan_kernel(const float* __restrict__ U,   const float* __restrict__ DBC,
                 const float* __restrict__ dtw, const float* __restrict__ dtb,
                 const float* __restrict__ Alog,const float* __restrict__ Dv,
                 const float* __restrict__ Zl,  float* __restrict__ Y) {
  const int tid   = threadIdx.x;
  const int b     = blockIdx.x >> 1;
  const int dbase = (blockIdx.x & 1) << 8;
  const int d     = dbase + tid;

  __shared__ float su[16][256];                 // U chunk      (16 KB)
  __shared__ float sdbc[16][80];                // dt+Bc chunk  (5 KB)

  float A[NN];
#pragma unroll
  for (int n2 = 0; n2 < NN; ++n2) A[n2] = -__expf(Alog[d * NN + n2]);
  float w[RR];
#pragma unroll
  for (int r = 0; r < RR; ++r) w[r] = dtw[d * RR + r];
  const float db = dtb[d];

  float hst[NN];
#pragma unroll
  for (int n2 = 0; n2 < NN; ++n2) hst[n2] = 0.f;

  const uint32_t su_base   = ldsoff(&su[0][0]);
  const uint32_t sdbc_base = ldsoff(&sdbc[0][0]);

  for (int t0 = 0; t0 < LL; t0 += 16) {
    __syncthreads();                            // prior chunk fully consumed

    // --- TDM: one wave issues the 2D dt/Bc tile DMA for the whole block ---
    if (tid < 32) {
      const uint64_t gaddr =
          (uint64_t)(uintptr_t)(DBC + (size_t)(b * LL + t0) * RC);
      u32x4 g0;
      g0[0] = 1u;                               // count=1, user descriptor
      g0[1] = sdbc_base;                        // lds_addr (bytes)
      g0[2] = (uint32_t)gaddr;                  // global_addr[31:0]
      g0[3] = (uint32_t)((gaddr >> 32) & 0x01FFFFFFu) | 0x80000000u; // +type=2
      u32x8 g1;
      g1[0] = 0x00020000u;                      // data_size=4B; no mask/pad
      g1[1] = 0u;                               // tensor_dim0[15:0]=0 in [31:16]
      g1[2] = 0x0010u;                          // tensor_dim0=0x100000 (no OOB)
      g1[3] = (80u << 16) | 0x0010u;            // tile_dim0=80 | tensor_dim1 hi
      g1[4] = 16u;                              // tile_dim1=16, tile_dim2=0
      g1[5] = (uint32_t)RC;                     // tensor_dim0_stride = 144
      g1[6] = 0u;
      g1[7] = 0u;
      asm volatile("tensor_load_to_lds %0, %1" :: "s"(g0), "s"(g1) : "memory");
    }

    // --- async per-lane DMA of the U chunk (coalesced across d) ---
    {
      const uint32_t l0   = su_base + (uint32_t)tid * 4u;
      const uint32_t goff =
          (uint32_t)(((size_t)(b * LL + t0) * DI + (size_t)d) * 4u);
#pragma unroll
      for (int tt = 0; tt < 16; ++tt)
        async_g2lds_b32(l0 + (uint32_t)(tt * 256 * 4),
                        goff + (uint32_t)(tt * DI * 4), U);
    }
    wait_asynccnt0();
    if (tid < 32) __builtin_amdgcn_s_wait_tensorcnt(0);
    __syncthreads();                            // staged data visible to all

    for (int tt = 0; tt < 16; ++tt) {
      const float u = su[tt][tid];
      float dp = db;
#pragma unroll
      for (int r = 0; r < RR; ++r) dp += sdbc[tt][r] * w[r];
      float delta = (dp > 20.f) ? dp : __logf(1.f + __expf(dp));  // softplus
      float du = delta * u;
#pragma unroll
      for (int n2 = 0; n2 < NN; ++n2) {
        float dA = __expf(delta * A[n2]);
        hst[n2] = dA * hst[n2] + du * sdbc[tt][16 + n2];
      }
    }
  }

  // epilogue: y = <h, C_last> + u_last*D, gated by silu(z_last)
  const float* Crow = DBC + (size_t)(b * LL + (LL - 1)) * RC + RR + NN;
  float y = 0.f;
#pragma unroll
  for (int n2 = 0; n2 < NN; ++n2) y += hst[n2] * Crow[n2];
  y += U[(size_t)(b * LL + (LL - 1)) * DI + d] * Dv[d];
  float z = Zl[b * DI + d];
  y *= z / (1.f + __expf(-z));
  Y[b * DI + d] = y;
}

// out_proj at last timestep: Feat(B,H) = Y(B,DI) @ ow^T + ob
__global__ __launch_bounds__(256)
void feat_kernel(const float* __restrict__ Y, const float* __restrict__ ow,
                 const float* __restrict__ ob, float* __restrict__ Feat) {
  int idx = blockIdx.x * 256 + threadIdx.x;     // B*H
  int b = idx >> 8, hh = idx & 255;
  const float* yr = Y + b * DI;
  const float* wrow = ow + hh * DI;
  float acc = ob[hh];
  for (int dd = 0; dd < DI; ++dd) acc += yr[dd] * wrow[dd];
  Feat[idx] = acc;
}

// fc1(relu) -> fc2 -> softmax, one block per batch element
__global__ __launch_bounds__(64)
void head_kernel(const float* __restrict__ Feat, const float* __restrict__ f1w,
                 const float* __restrict__ f1b,  const float* __restrict__ f2w,
                 const float* __restrict__ f2b,  float* __restrict__ Out) {
  int b = blockIdx.x;
  __shared__ float h1[64];
  const float* fr = Feat + b * HH;
  const float* wrow = f1w + threadIdx.x * HH;
  float acc = f1b[threadIdx.x];
  for (int h = 0; h < HH; ++h) acc += fr[h] * wrow[h];
  h1[threadIdx.x] = fmaxf(acc, 0.f);
  __syncthreads();
  if (threadIdx.x == 0) {
    float l0 = f2b[0], l1 = f2b[1];
    for (int j = 0; j < 64; ++j) { l0 += h1[j] * f2w[j]; l1 += h1[j] * f2w[64 + j]; }
    float mx = fmaxf(l0, l1);
    float e0 = __expf(l0 - mx), e1 = __expf(l1 - mx);
    float s = e0 + e1;
    Out[b * 2 + 0] = e0 / s;
    Out[b * 2 + 1] = e1 / s;
  }
}

extern "C" void kernel_launch(void* const* d_in, const int* in_sizes, int n_in,
                              void* d_out, int out_size, void* d_ws, size_t ws_size,
                              hipStream_t stream) {
  const float* x    = (const float*)d_in[0];
  const float* ew   = (const float*)d_in[1];
  const float* eb   = (const float*)d_in[2];
  const float* ipw  = (const float*)d_in[3];
  const float* ipb  = (const float*)d_in[4];
  const float* cw   = (const float*)d_in[5];
  const float* cb   = (const float*)d_in[6];
  const float* xpw  = (const float*)d_in[7];
  const float* dtw  = (const float*)d_in[8];
  const float* dtb  = (const float*)d_in[9];
  const float* alog = (const float*)d_in[10];
  const float* Dv   = (const float*)d_in[11];
  const float* ow   = (const float*)d_in[12];
  const float* ob   = (const float*)d_in[13];
  const float* f1w  = (const float*)d_in[14];
  const float* f1b  = (const float*)d_in[15];
  const float* f2w  = (const float*)d_in[16];
  const float* f2b  = (const float*)d_in[17];

  float* ws = (float*)d_ws;
  float* Wc   = ws;               // F*DI        = 10240
  float* Bcb  = Wc   + FF * DI;   // DI
  float* Hl   = Bcb  + DI;        // B*H
  float* Zl   = Hl   + BB * HH;   // B*DI
  float* Yv   = Zl   + BB * DI;   // B*DI
  float* Feat = Yv   + BB * DI;   // B*H
  float* Upre = Feat + BB * HH;   // BL*DI
  float* Uc   = Upre + (size_t)BL * DI;   // BL*DI
  float* Dbc  = Uc   + (size_t)BL * DI;   // BL*RC
  // total ~9.63M floats (~38.5 MB)

  prep_kernel     <<<(FF * DI + DI + 255) / 256, 256, 0, stream>>>(ew, eb, ipw, ipb, Wc, Bcb);
  hlast_kernel    <<<BB, 256, 0, stream>>>(x, ew, eb, Hl);
  zlast_kernel    <<<(BB * DI) / 256, 256, 0, stream>>>(Hl, ipw, ipb, Zl);
  gemm_upre_wmma  <<<(BL / 16) * (DI / 16) / 8, 256, 0, stream>>>(x, Wc, Bcb, Upre);
  conv_silu_kernel<<<(BL * DI) / 256, 256, 0, stream>>>(Upre, cw, cb, Uc);
  gemm_dbc_wmma   <<<(BL / 16) * (RC / 16) / 8, 256, 0, stream>>>(Uc, xpw, Dbc);
  scan_kernel     <<<BB * 2, 256, 0, stream>>>(Uc, Dbc, dtw, dtb, alog, Dv, Zl, Yv);
  feat_kernel     <<<BB, 256, 0, stream>>>(Yv, ow, ob, Feat);
  head_kernel     <<<BB, 64, 0, stream>>>(Feat, f1w, f1b, f2w, f2b, (float*)d_out);
}
